// MLPSparseMoe_63660005262031
// MI455X (gfx1250) — matrix operand order repaired and checked by reading, not compiled, not run on hardware
//
#include <hip/hip_runtime.h>
#include <hip/hip_bf16.h>
#include <math.h>

typedef __bf16 bf16_t;
typedef __attribute__((ext_vector_type(16))) __bf16 v16bf;
typedef __attribute__((ext_vector_type(8)))  __bf16 v8bf;
typedef __attribute__((ext_vector_type(4)))  __bf16 v4bf;
typedef __attribute__((ext_vector_type(8)))  float  v8f;
typedef __attribute__((ext_vector_type(4)))  float  v4f;

#define T_TOK 8192
#define DDIM  1024
#define HDIM  4096
#define NEXP  8

// ---------------------------------------------------------------------------
// Zero-init y and expert counts.
// ---------------------------------------------------------------------------
__global__ void moe_zero_kernel(float* __restrict__ y, int n4, int* __restrict__ counts) {
    int i = blockIdx.x * blockDim.x + threadIdx.x;
    if (i < n4) {
        v4f z; z[0] = 0.f; z[1] = 0.f; z[2] = 0.f; z[3] = 0.f;
        ((v4f*)y)[i] = z;
    }
    if (blockIdx.x == 0 && threadIdx.x < NEXP) counts[threadIdx.x] = 0;
}

// ---------------------------------------------------------------------------
// x (f32) -> xb (bf16), elementwise, 4 at a time.
// ---------------------------------------------------------------------------
__global__ void moe_cvt_x_kernel(const float* __restrict__ x, bf16_t* __restrict__ xb, int n4) {
    int i = blockIdx.x * blockDim.x + threadIdx.x;
    if (i >= n4) return;
    v4f v = ((const v4f*)x)[i];
    v4bf o;
    o[0] = (bf16_t)v[0]; o[1] = (bf16_t)v[1]; o[2] = (bf16_t)v[2]; o[3] = (bf16_t)v[3];
    ((v4bf*)xb)[i] = o;
}

// ---------------------------------------------------------------------------
// Batched transpose + f32->bf16: src[b][R][C] f32  ->  dst[b][C][R] bf16.
// 32x32 LDS tile, 256 threads (32x8).
// ---------------------------------------------------------------------------
__global__ __launch_bounds__(256) void moe_transpose_cvt_kernel(
    const float* __restrict__ src, bf16_t* __restrict__ dst, int R, int C) {
    __shared__ float tile[32][33];
    const int b  = blockIdx.z;
    const float* s = src + (size_t)b * R * C;
    bf16_t*      d = dst + (size_t)b * R * C;
    const int tx = threadIdx.x & 31;
    const int ty = threadIdx.x >> 5;
    const int c0 = blockIdx.x * 32;
    const int r0 = blockIdx.y * 32;
#pragma unroll
    for (int i = 0; i < 4; ++i) {
        int r = r0 + ty + i * 8;
        tile[ty + i * 8][tx] = s[(size_t)r * C + (c0 + tx)];
    }
    __syncthreads();
#pragma unroll
    for (int i = 0; i < 4; ++i) {
        int c = c0 + ty + i * 8;
        d[(size_t)c * R + (r0 + tx)] = (bf16_t)tile[tx][ty + i * 8];
    }
}

// ---------------------------------------------------------------------------
// Router: one wave32 per token. logits = x @ Wg^T (raw, written to output),
// top-2 + renormalized softmax weights, build per-expert pair lists.
// ---------------------------------------------------------------------------
__global__ __launch_bounds__(128) void moe_router_kernel(
    const float* __restrict__ x, const float* __restrict__ Wg,
    float* __restrict__ logits_out, int* __restrict__ counts,
    int* __restrict__ plist, float* __restrict__ pw) {
    const int wave = threadIdx.x >> 5;
    const int lane = threadIdx.x & 31;
    const int t = blockIdx.x * 4 + wave;
    if (t >= T_TOK) return;

    float acc[NEXP];
#pragma unroll
    for (int e = 0; e < NEXP; ++e) acc[e] = 0.f;
    const float* xr = x + (size_t)t * DDIM;
    for (int d = lane; d < DDIM; d += 32) {
        float xv = xr[d];
#pragma unroll
        for (int e = 0; e < NEXP; ++e) acc[e] += xv * Wg[e * DDIM + d];
    }
#pragma unroll
    for (int e = 0; e < NEXP; ++e) {
#pragma unroll
        for (int off = 16; off > 0; off >>= 1)
            acc[e] += __shfl_xor(acc[e], off, 32);
    }
    if (lane == 0) {
#pragma unroll
        for (int e = 0; e < NEXP; ++e) logits_out[(size_t)t * NEXP + e] = acc[e];
        // top-1
        int e0 = 0; float v0 = acc[0];
#pragma unroll
        for (int e = 1; e < NEXP; ++e)
            if (acc[e] > v0) { v0 = acc[e]; e0 = e; }
        // top-2
        int e1 = (e0 == 0) ? 1 : 0; float v1 = acc[e1];
#pragma unroll
        for (int e = 0; e < NEXP; ++e)
            if (e != e0 && acc[e] > v1) { v1 = acc[e]; e1 = e; }
        // renormalized softmax over the two picks (exact)
        float w0 = 1.0f / (1.0f + expf(v1 - v0));
        float w1 = 1.0f - w0;
        int p0 = atomicAdd(&counts[e0], 1);
        plist[e0 * T_TOK + p0] = t * 2 + 0;
        pw[e0 * T_TOK + p0]    = w0;
        int p1 = atomicAdd(&counts[e1], 1);
        plist[e1 * T_TOK + p1] = t * 2 + 1;
        pw[e1 * T_TOK + p1]    = w1;
    }
}

// ---------------------------------------------------------------------------
// Fragment helper: 16 bf16 from two 16B chunks at p and p + hiOfs.
// ---------------------------------------------------------------------------
static __device__ __forceinline__ v16bf load_frag(const bf16_t* p, int hiOfs) {
    v8bf lo = *(const v8bf*)(p);
    v8bf hi = *(const v8bf*)(p + hiOfs);
    return __builtin_shufflevector(lo, hi, 0, 1, 2, 3, 4, 5, 6, 7,
                                           8, 9, 10, 11, 12, 13, 14, 15);
}

// ---------------------------------------------------------------------------
// Up-projection grouped GEMM: h[pair] = gelu(x[token] @ W1[e]).
// A gathered from xb via per-expert pair list; B from W1t ([H][D] bf16).
// Block = 128 threads = 4 waves (2x2), each wave 64x64 (4x4 WMMA frags).
// ---------------------------------------------------------------------------
__global__ __launch_bounds__(128) void moe_up_kernel(
    const bf16_t* __restrict__ xb, const bf16_t* __restrict__ w1t,
    const int* __restrict__ counts, const int* __restrict__ plist,
    bf16_t* __restrict__ h) {
    const int e   = blockIdx.z;
    const int cnt = counts[e];
    const int rowBase = blockIdx.y * 128;
    if (rowBase >= cnt) return;

    const int lane   = threadIdx.x & 31;
    const int wave   = threadIdx.x >> 5;
    const int wm     = rowBase + (wave >> 1) * 64;
    const int wn     = blockIdx.x * 128 + (wave & 1) * 64;
    const int l15    = lane & 15;
    const int hiHalf = lane >> 4;
    const int kOffA  = hiHalf * 8;    // A: lane half-runs at +0/+8, second run +16
    const int kOffB  = hiHalf * 8;    // B: lane half-runs at +0/+8, second run +16
    const int* lst   = plist + e * T_TOK;

    const bf16_t* aPtr[4];
#pragma unroll
    for (int i = 0; i < 4; ++i) {
        int m  = wm + i * 16 + l15;
        int mm = (m < cnt) ? m : (cnt - 1);
        int pid = lst[mm];
        aPtr[i] = xb + (size_t)(pid >> 1) * DDIM + kOffA;
    }
    const bf16_t* bPtr[4];
#pragma unroll
    for (int j = 0; j < 4; ++j) {
        int n = wn + j * 16 + l15;
        bPtr[j] = w1t + ((size_t)e * HDIM + n) * DDIM + kOffB;
    }

    v8f acc[4][4];
#pragma unroll
    for (int i = 0; i < 4; ++i)
#pragma unroll
        for (int j = 0; j < 4; ++j)
#pragma unroll
            for (int k = 0; k < 8; ++k) acc[i][j][k] = 0.f;

    for (int kb = 0; kb < DDIM; kb += 32) {
        v16bf a[4], b[4];
#pragma unroll
        for (int i = 0; i < 4; ++i) a[i] = load_frag(aPtr[i] + kb, 16);
#pragma unroll
        for (int j = 0; j < 4; ++j) b[j] = load_frag(bPtr[j] + kb, 16);
#pragma unroll
        for (int i = 0; i < 4; ++i)
#pragma unroll
            for (int j = 0; j < 4; ++j)
                acc[i][j] = __builtin_amdgcn_wmma_f32_16x16x32_bf16(
                    false, a[i], false, b[j], (short)0, acc[i][j], false, false);
    }

    // exact GELU + store h (bf16), rows guarded by count
#pragma unroll
    for (int i = 0; i < 4; ++i) {
#pragma unroll
        for (int r = 0; r < 8; ++r) {
            int m = wm + i * 16 + hiHalf * 8 + r;
            if (m < cnt) {
                int pid = lst[m];
                bf16_t* hrow = h + (size_t)pid * HDIM;
#pragma unroll
                for (int j = 0; j < 4; ++j) {
                    float v = acc[i][j][r];
                    float g = 0.5f * v * (1.0f + erff(v * 0.70710678118654752f));
                    hrow[wn + j * 16 + l15] = (bf16_t)g;
                }
            }
        }
    }
}

// ---------------------------------------------------------------------------
// Down-projection grouped GEMM: y[token] += w_pair * (h[pair] @ W2[e]).
// Row gating commutes with GEMM -> applied at output; f32 global atomics.
// ---------------------------------------------------------------------------
__global__ __launch_bounds__(128) void moe_down_kernel(
    const bf16_t* __restrict__ h, const bf16_t* __restrict__ w2t,
    const int* __restrict__ counts, const int* __restrict__ plist,
    const float* __restrict__ pw, float* __restrict__ y) {
    const int e   = blockIdx.z;
    const int cnt = counts[e];
    const int rowBase = blockIdx.y * 128;
    if (rowBase >= cnt) return;

    const int lane   = threadIdx.x & 31;
    const int wave   = threadIdx.x >> 5;
    const int wm     = rowBase + (wave >> 1) * 64;
    const int wn     = blockIdx.x * 128 + (wave & 1) * 64;
    const int l15    = lane & 15;
    const int hiHalf = lane >> 4;
    const int kOffA  = hiHalf * 8;
    const int kOffB  = hiHalf * 8;
    const int* lst   = plist + e * T_TOK;

    const bf16_t* aPtr[4];
#pragma unroll
    for (int i = 0; i < 4; ++i) {
        int m  = wm + i * 16 + l15;
        int mm = (m < cnt) ? m : (cnt - 1);
        int pid = lst[mm];
        aPtr[i] = h + (size_t)pid * HDIM + kOffA;
    }
    const bf16_t* bPtr[4];
#pragma unroll
    for (int j = 0; j < 4; ++j) {
        int n = wn + j * 16 + l15;
        bPtr[j] = w2t + ((size_t)e * DDIM + n) * HDIM + kOffB;
    }

    v8f acc[4][4];
#pragma unroll
    for (int i = 0; i < 4; ++i)
#pragma unroll
        for (int j = 0; j < 4; ++j)
#pragma unroll
            for (int k = 0; k < 8; ++k) acc[i][j][k] = 0.f;

    for (int kb = 0; kb < HDIM; kb += 32) {
        v16bf a[4], b[4];
#pragma unroll
        for (int i = 0; i < 4; ++i) a[i] = load_frag(aPtr[i] + kb, 16);
#pragma unroll
        for (int j = 0; j < 4; ++j) b[j] = load_frag(bPtr[j] + kb, 16);
#pragma unroll
        for (int i = 0; i < 4; ++i)
#pragma unroll
            for (int j = 0; j < 4; ++j)
                acc[i][j] = __builtin_amdgcn_wmma_f32_16x16x32_bf16(
                    false, a[i], false, b[j], (short)0, acc[i][j], false, false);
    }

#pragma unroll
    for (int i = 0; i < 4; ++i) {
#pragma unroll
        for (int r = 0; r < 8; ++r) {
            int m = wm + i * 16 + hiHalf * 8 + r;
            if (m < cnt) {
                int   pid = lst[m];
                float w   = pw[e * T_TOK + m];
                float* yrow = y + (size_t)(pid >> 1) * DDIM;
#pragma unroll
                for (int j = 0; j < 4; ++j)
                    atomicAdd(yrow + (wn + j * 16 + l15), w * acc[i][j][r]);
            }
        }
    }
}

// ---------------------------------------------------------------------------
// Launch
// ---------------------------------------------------------------------------
extern "C" void kernel_launch(void* const* d_in, const int* in_sizes, int n_in,
                              void* d_out, int out_size, void* d_ws, size_t ws_size,
                              hipStream_t stream) {
    const float* x  = (const float*)d_in[0];
    const float* Wg = (const float*)d_in[1];
    const float* W1 = (const float*)d_in[2];
    const float* W2 = (const float*)d_in[3];

    float* y      = (float*)d_out;
    float* logits = y + (size_t)T_TOK * DDIM;

    char* ws = (char*)d_ws;
    size_t off = 0;
    auto take = [&](size_t bytes) -> char* {
        char* p = ws + off;
        off = (off + bytes + 255) & ~(size_t)255;
        return p;
    };
    int*    counts = (int*)take(NEXP * sizeof(int));
    int*    plist  = (int*)take((size_t)NEXP * T_TOK * sizeof(int));
    float*  pwts   = (float*)take((size_t)NEXP * T_TOK * sizeof(float));
    bf16_t* xb     = (bf16_t*)take((size_t)T_TOK * DDIM * sizeof(bf16_t));
    bf16_t* w1t    = (bf16_t*)take((size_t)NEXP * DDIM * HDIM * sizeof(bf16_t));
    bf16_t* w2t    = (bf16_t*)take((size_t)NEXP * DDIM * HDIM * sizeof(bf16_t));
    bf16_t* hbuf   = (bf16_t*)take((size_t)T_TOK * 2 * HDIM * sizeof(bf16_t));
    (void)ws_size; (void)in_sizes; (void)n_in; (void)out_size;

    {
        int n4 = T_TOK * DDIM / 4;
        moe_zero_kernel<<<(n4 + 255) / 256, 256, 0, stream>>>(y, n4, counts);
        moe_cvt_x_kernel<<<(n4 + 255) / 256, 256, 0, stream>>>(x, xb, n4);
    }
    {
        dim3 g(HDIM / 32, DDIM / 32, NEXP);   // W1: [D][H] -> [H][D]
        moe_transpose_cvt_kernel<<<g, 256, 0, stream>>>(W1, w1t, DDIM, HDIM);
    }
    {
        dim3 g(DDIM / 32, HDIM / 32, NEXP);   // W2: [H][D] -> [D][H]
        moe_transpose_cvt_kernel<<<g, 256, 0, stream>>>(W2, w2t, HDIM, DDIM);
    }
    moe_router_kernel<<<T_TOK / 4, 128, 0, stream>>>(x, Wg, logits, counts, plist, pwts);
    {
        dim3 g(HDIM / 128, T_TOK / 128, NEXP);
        moe_up_kernel<<<g, 128, 0, stream>>>(xb, w1t, counts, plist, hbuf);
    }
    {
        dim3 g(DDIM / 128, T_TOK / 128, NEXP);
        moe_down_kernel<<<g, 128, 0, stream>>>(hbuf, w2t, counts, plist, pwts, y);
    }
}
